// KQVPCNN_73194832659002
// MI455X (gfx1250) — matrix operand order repaired and checked by reading, not compiled, not run on hardware
//
#include <hip/hip_runtime.h>

// ---------------------------------------------------------------------------
// Types for WMMA fragments (CDNA5 wave32: v_wmma_f32_16x16x32_bf16)
// ---------------------------------------------------------------------------
typedef __attribute__((ext_vector_type(16))) __bf16 bf16x16;
typedef __attribute__((ext_vector_type(8)))  float  f32x8;
typedef int v4i_vs __attribute__((vector_size(16)));   // matches async-LDS builtin param

struct alignas(16) U4 { unsigned x, y, z, w; };
union Frag { bf16x16 v; U4 q[2]; };        // 32 bytes = 16 bf16 per lane

static __device__ __forceinline__ unsigned short f2bf(float x) {
    union { float f; unsigned u; } c; c.f = x;
    unsigned u = c.u;
    u += 0x7fffu + ((u >> 16) & 1u);       // round-to-nearest-even
    return (unsigned short)(u >> 16);
}

#define WMMA_BF16(A, B, C) \
    __builtin_amdgcn_wmma_f32_16x16x32_bf16(false, (A), false, (B), (short)0, (C), false, false)

// ---- async global->LDS copy (CDNA5), with safe fallback --------------------
#ifndef __has_builtin
#define __has_builtin(x) 0
#endif

#if __has_builtin(__builtin_amdgcn_global_load_async_to_lds_b128)
#define HAS_ASYNC 1
#define ASYNC_CP(g, l)                                                     \
    __builtin_amdgcn_global_load_async_to_lds_b128(                        \
        (__attribute__((address_space(1))) v4i_vs*)(g),                    \
        (__attribute__((address_space(3))) v4i_vs*)(l), 0, 0)
#else
#define HAS_ASYNC 0
#define ASYNC_CP(g, l) (*(U4*)(l) = *(const U4*)(g))
#endif

#if HAS_ASYNC
#if __has_builtin(__builtin_amdgcn_s_wait_asynccnt)
#define WAIT_ASYNC() __builtin_amdgcn_s_wait_asynccnt(0)
#else
#define WAIT_ASYNC() asm volatile("s_wait_asynccnt 0" ::: "memory")
#endif
#else
#define WAIT_ASYNC() ((void)0)
#endif

static constexpr int Bn   = 4;
static constexpr int S    = 4096;
static constexpr int D    = 256;
static constexpr int KC   = 768;           // im2col depth = 3*256

// ---------------------------------------------------------------------------
// Kernel 0: repack conv weights f32 [O][I][3] -> bf16 [O][k*256+i]
// ---------------------------------------------------------------------------
__global__ __launch_bounds__(256) void wrepack_kernel(
    const float* __restrict__ qw, const float* __restrict__ kw,
    const float* __restrict__ vw, unsigned short* __restrict__ out)
{
    const int per = D * KC;                                // 196608
    int t = blockIdx.x * 256 + threadIdx.x;
    if (t >= 3 * per) return;
    int p = t / per, rr = t - p * per;
    int o = rr / KC, c = rr - o * KC;
    int k = c >> 8, i = c & 255;
    const float* w = (p == 0) ? qw : ((p == 1) ? kw : vw);
    out[t] = f2bf(w[o * KC + i * 3 + k]);
}

// ---------------------------------------------------------------------------
// Kernel 1: conv1d(K=3,pad=1) as GEMM via WMMA.
//   grid = (BS/64, 256/64, 3), block = 128 (4 waves).
//   z==0 -> Q row-major bf16; z==1 -> K row-major; z==2 -> V transposed [B][D][S].
// ---------------------------------------------------------------------------
__global__ __launch_bounds__(128) void convqkv_kernel(
    const float* __restrict__ xq, const float* __restrict__ xk, const float* __restrict__ xv,
    const float* __restrict__ bq, const float* __restrict__ bk, const float* __restrict__ bv,
    const unsigned short* __restrict__ wc,
    unsigned short* __restrict__ Qb, unsigned short* __restrict__ Kb,
    unsigned short* __restrict__ Vt)
{
    __shared__ alignas(16) unsigned short xs[66 * 256];    // 33 KB staged activations (bf16)

    const int z     = blockIdx.z;
    const float* x    = (z == 0) ? xq : ((z == 1) ? xk : xv);
    const float* bias = (z == 0) ? bq : ((z == 1) ? bk : bv);
    const unsigned short* w = wc + (size_t)z * D * KC;

    const int s0    = blockIdx.x * 64;                     // flat (b*S+s) row base
    const int batch = s0 >> 12;
    const int lo    = batch << 12;
    const int o0    = blockIdx.y * 64;
    const int tid   = threadIdx.x;

    // stage rows s0-1 .. s0+64 (zero-padded at batch edges), f32 -> bf16
    for (int it = 0; it < 33; ++it) {
        int idx4 = it * 128 + tid;                          // 4224 float4's
        int row  = idx4 >> 6;                               // 64 float4 per row
        int col  = (idx4 & 63) << 2;
        int s    = s0 - 1 + row;
        float4 val = make_float4(0.f, 0.f, 0.f, 0.f);
        if (s >= lo && s < lo + S) val = *(const float4*)(x + (size_t)s * D + col);
        unsigned p0 = (unsigned)f2bf(val.x) | ((unsigned)f2bf(val.y) << 16);
        unsigned p1 = (unsigned)f2bf(val.z) | ((unsigned)f2bf(val.w) << 16);
        *(uint2*)(&xs[row * 256 + col]) = make_uint2(p0, p1);
    }
    __syncthreads();

    const int wid  = tid >> 5;
    const int lane = tid & 31;
    const int r    = lane & 15;
    const int h    = lane >> 4;

    f32x8 acc[4];
#pragma unroll
    for (int nt = 0; nt < 4; ++nt) {
        float bv0 = bias[o0 + nt * 16 + r];
#pragma unroll
        for (int j = 0; j < 8; ++j) acc[nt][j] = bv0;
    }

    for (int c0 = 0; c0 < KC; c0 += 32) {
        const int kk = c0 >> 8;                             // conv tap (const per chunk)
        const int i0 = c0 & 255;
        Frag a;
        const unsigned short* arow = &xs[(wid * 16 + r + kk) * 256 + i0];
        a.q[0] = *(const U4*)(arow + 8 * h);
        a.q[1] = *(const U4*)(arow + 16 + 8 * h);
#pragma unroll
        for (int nt = 0; nt < 4; ++nt) {
            Frag bf;
            const unsigned short* brow = w + (size_t)(o0 + nt * 16 + r) * KC + c0 + 8 * h;
            bf.q[0] = *(const U4*)(brow);
            bf.q[1] = *(const U4*)(brow + 16);
            acc[nt] = WMMA_BF16(a.v, bf.v, acc[nt]);
        }
    }

    if (z < 2) {
        unsigned short* out = (z == 0) ? Qb : Kb;
#pragma unroll
        for (int nt = 0; nt < 4; ++nt) {
            int o = o0 + nt * 16 + r;
#pragma unroll
            for (int j = 0; j < 8; ++j) {
                int srow = s0 + wid * 16 + 8 * h + j;
                out[(size_t)srow * D + o] = f2bf(acc[nt][j]);
            }
        }
    } else {
        // V transposed: Vt[batch][o][s], 8 consecutive s per lane -> one b128 store
#pragma unroll
        for (int nt = 0; nt < 4; ++nt) {
            int o  = o0 + nt * 16 + r;
            int sb = (s0 - lo) + wid * 16 + 8 * h;
            union { U4 q; unsigned short u[8]; } pk;
#pragma unroll
            for (int j = 0; j < 8; ++j) pk.u[j] = f2bf(acc[nt][j]);
            *(U4*)(Vt + ((size_t)batch * D + o) * S + sb) = pk.q;
        }
    }
}

// ---------------------------------------------------------------------------
// Kernel 2: flash attention. grid = (S/64, B), block = 128 (4 waves).
// K/V chunks (64 keys) cooperatively staged to LDS (async, double-buffered);
// each wave: 16 queries, d=256 output held in 128 f32 accumulator VGPRs.
// ---------------------------------------------------------------------------
__global__ __launch_bounds__(128) void attn_kernel(
    const unsigned short* __restrict__ Qb, const unsigned short* __restrict__ Kb,
    const unsigned short* __restrict__ Vt, const float* __restrict__ scale_p,
    float* __restrict__ out)
{
    __shared__ alignas(16) unsigned short kbuf[2][64 * 256];   // 2 x 32 KB
    __shared__ alignas(16) unsigned short vbuf[2][256 * 64];   // 2 x 32 KB
    __shared__ alignas(16) unsigned short psh[4 * 16 * 64];    // 2 KB per wave

    const int b    = blockIdx.y;
    const int tid  = threadIdx.x;
    const int wid  = tid >> 5;
    const int lane = tid & 31;
    const int r    = lane & 15;
    const int h    = lane >> 4;
    const int q0   = blockIdx.x * 64 + wid * 16;
    unsigned short* pw = &psh[wid * 1024];

    const unsigned short* Kg0 = Kb + (size_t)b * S * D;        // [S][256]
    const unsigned short* Vg0 = Vt + (size_t)b * D * S;        // [256][S]

    // Q fragments resident in registers (16 queries x 256 d = 8 frags)
    Frag qf[8];
    const unsigned short* qrow = Qb + (size_t)(b * S + q0 + r) * D;
#pragma unroll
    for (int f = 0; f < 8; ++f) {
        qf[f].q[0] = *(const U4*)(qrow + f * 32 + 8 * h);
        qf[f].q[1] = *(const U4*)(qrow + f * 32 + 16 + 8 * h);
    }

    f32x8 acc[16];
#pragma unroll
    for (int dt = 0; dt < 16; ++dt)
#pragma unroll
        for (int j = 0; j < 8; ++j) acc[dt][j] = 0.f;

    float mrow[8], lrow[8];
#pragma unroll
    for (int j = 0; j < 8; ++j) { mrow[j] = -1e30f; lrow[j] = 0.f; }

    const float iscl = 0.0625f;                                // 1/sqrt(256)

    // ---- cooperative stage of one 64-key chunk: K 32KB + V 32KB ----------
    auto stage = [&](int buf, int kc) {
        unsigned short* kl = kbuf[buf];
        unsigned short* vl = vbuf[buf];
        const unsigned short* kg = Kg0 + (size_t)kc * D;       // contiguous 32 KB
#pragma unroll
        for (int i = 0; i < 16; ++i) {                          // 2048 x 16B, 128 thr
            int e = i * 128 + tid;
            ASYNC_CP(kg + e * 8, kl + e * 8);
        }
#pragma unroll
        for (int i = 0; i < 16; ++i) {                          // 256 rows x 8 x 16B
            int e = i * 128 + tid;
            int d = e >> 3, p = e & 7;
            ASYNC_CP(Vg0 + (size_t)d * S + kc + p * 8, vl + d * 64 + p * 8);
        }
    };

    stage(0, 0);
    WAIT_ASYNC();
    __syncthreads();

    const int nChunk = S / 64;
    for (int c = 0; c < nChunk; ++c) {
        const int cur = c & 1;
        if (c + 1 < nChunk) stage(cur ^ 1, (c + 1) * 64);       // overlap DMA w/ WMMA

        const unsigned short* kl = kbuf[cur];
        const unsigned short* vl = vbuf[cur];

        // ---- scores: S = Q K^T, 4 n-tiles of 16 keys --------------------
        f32x8 st[4];
#pragma unroll
        for (int nt = 0; nt < 4; ++nt) {
            f32x8 cc;
#pragma unroll
            for (int j = 0; j < 8; ++j) cc[j] = 0.f;
            const unsigned short* krow = kl + (nt * 16 + r) * 256;
#pragma unroll
            for (int f = 0; f < 8; ++f) {
                Frag kf;
                kf.q[0] = *(const U4*)(krow + f * 32 + 8 * h);
                kf.q[1] = *(const U4*)(krow + f * 32 + 16 + 8 * h);
                cc = WMMA_BF16(qf[f].v, kf.v, cc);
            }
            st[nt] = cc;
        }

        // ---- online softmax (rows live in 16-lane halves) ----------------
        float curm[8];
#pragma unroll
        for (int j = 0; j < 8; ++j) {
            st[0][j] *= iscl; st[1][j] *= iscl; st[2][j] *= iscl; st[3][j] *= iscl;
            curm[j] = fmaxf(fmaxf(st[0][j], st[1][j]), fmaxf(st[2][j], st[3][j]));
        }
#pragma unroll
        for (int m = 1; m < 16; m <<= 1)
#pragma unroll
            for (int j = 0; j < 8; ++j) curm[j] = fmaxf(curm[j], __shfl_xor(curm[j], m, 32));

        float corr[8], rs[8];
#pragma unroll
        for (int j = 0; j < 8; ++j) {
            float nm = fmaxf(mrow[j], curm[j]);
            corr[j]  = __expf(mrow[j] - nm);
            mrow[j]  = nm;
            rs[j]    = 0.f;
        }
#pragma unroll
        for (int nt = 0; nt < 4; ++nt)
#pragma unroll
            for (int j = 0; j < 8; ++j) {
                float p = __expf(st[nt][j] - mrow[j]);
                st[nt][j] = p;
                rs[j] += p;
            }
#pragma unroll
        for (int m = 1; m < 16; m <<= 1)
#pragma unroll
            for (int j = 0; j < 8; ++j) rs[j] += __shfl_xor(rs[j], m, 32);
#pragma unroll
        for (int j = 0; j < 8; ++j) lrow[j] = lrow[j] * corr[j] + rs[j];
#pragma unroll
        for (int dt = 0; dt < 16; ++dt)
#pragma unroll
            for (int j = 0; j < 8; ++j) acc[dt][j] *= corr[j];

        // ---- transpose P (C-layout) -> A-fragment layout via LDS ---------
#pragma unroll
        for (int nt = 0; nt < 4; ++nt)
#pragma unroll
            for (int j = 0; j < 8; ++j)
                pw[(8 * h + j) * 64 + nt * 16 + r] = f2bf(st[nt][j]);

        // ---- P.V from LDS-staged transposed V ----------------------------
#pragma unroll
        for (int sc2 = 0; sc2 < 2; ++sc2) {
            Frag pa;
            const unsigned short* parow = pw + r * 64 + sc2 * 32;
            pa.q[0] = *(const U4*)(parow + 8 * h);
            pa.q[1] = *(const U4*)(parow + 16 + 8 * h);
#pragma unroll
            for (int dt = 0; dt < 16; ++dt) {
                Frag vf;
                const unsigned short* vrow = vl + (dt * 16 + r) * 64 + sc2 * 32 + 8 * h;
                vf.q[0] = *(const U4*)(vrow);
                vf.q[1] = *(const U4*)(vrow + 16);
                acc[dt] = WMMA_BF16(pa.v, vf.v, acc[dt]);
            }
        }

        WAIT_ASYNC();            // my slice of next chunk landed in LDS
        __syncthreads();         // everyone done reading cur + done staging next
    }

    const float sc = scale_p[0];
#pragma unroll
    for (int dt = 0; dt < 16; ++dt) {
        int d = dt * 16 + r;
#pragma unroll
        for (int j = 0; j < 8; ++j) {
            int q = q0 + 8 * h + j;
            out[(size_t)(b * S + q) * D + d] = acc[dt][j] * (sc / lrow[j]);
        }
    }
}

// ---------------------------------------------------------------------------
// Launch
// ---------------------------------------------------------------------------
extern "C" void kernel_launch(void* const* d_in, const int* in_sizes, int n_in,
                              void* d_out, int out_size, void* d_ws, size_t ws_size,
                              hipStream_t stream)
{
    const float* xq = (const float*)d_in[0];
    const float* xk = (const float*)d_in[1];
    const float* xv = (const float*)d_in[2];
    const float* qw = (const float*)d_in[3];
    const float* qb = (const float*)d_in[4];
    const float* kw = (const float*)d_in[5];
    const float* kb = (const float*)d_in[6];
    const float* vw = (const float*)d_in[7];
    const float* vb = (const float*)d_in[8];
    const float* sc = (const float*)d_in[9];

    unsigned short* wc = (unsigned short*)d_ws;            // [3][256][768] bf16
    unsigned short* Qb = wc + (size_t)3 * D * KC;          // [B*S][256] bf16
    unsigned short* Kb = Qb + (size_t)Bn * S * D;
    unsigned short* Vt = Kb + (size_t)Bn * S * D;          // [B][256][S] bf16

    wrepack_kernel<<<(3 * D * KC + 255) / 256, 256, 0, stream>>>(qw, kw, vw, wc);

    dim3 g1(Bn * S / 64, D / 64, 3);
    convqkv_kernel<<<g1, 128, 0, stream>>>(xq, xk, xv, qb, kb, vb, wc, Qb, Kb, Vt);

    dim3 g2(S / 64, Bn);
    attn_kernel<<<g2, 128, 0, stream>>>(Qb, Kb, Vt, sc, (float*)d_out);
}
